// GroupWiseCostVolume3D_91199335563539
// MI455X (gfx1250) — matrix operand order repaired and checked by reading, not compiled, not run on hardware
//
#include <hip/hip_runtime.h>

// ---------------------------------------------------------------------------
// GroupWiseCostVolume3D for MI455X (gfx1250, wave32, WMMA).
//
// Pipeline per dilation:
//   corr_kernel : grouped 81-disp correlation (fp32) + leaky-ReLU -> bf16
//                 cost volume in ws, pixel-major [b][disp][h][w][g16].
//   conv_kernel : 3x3x3 conv as 16x16 GEMM tiles on V_WMMA_F32_16X16X32_BF16,
//                 K = (kw,g_in) split into two K=32 slices per (kd,kh)
//                 (kw padded 3->4 with zero weights). Halo tile staged into
//                 LDS with GLOBAL_LOAD_ASYNC_TO_LDS_B128 (ASYNCcnt path).
//                 BN+bias+ReLU epilogue, non-temporal f32 stores.
// prep_kernel   : one-time swizzle of weights into a contiguous-k (K=16*half+e) A-matrix
//                 per-lane fragment layout so the hot loop does contiguous
//                 32B fragment loads.
// ---------------------------------------------------------------------------

typedef float  v8f   __attribute__((ext_vector_type(8)));
typedef __bf16 v16bf __attribute__((ext_vector_type(16)));

union Frag32 { uint4 q[2]; v16bf v; };

__device__ __forceinline__ unsigned short f2bf(float f) {
  unsigned u = __float_as_uint(f);
  unsigned r = u + 0x7FFFu + ((u >> 16) & 1u);   // round-to-nearest-even
  return (unsigned short)(r >> 16);
}

#define NB 2
#define NC 128
#define NH 64
#define NW 64
#define NG 16
#define ND 81          // 9x9 displacements

// ---------------------------------------------------------------------------
// Correlation: one thread per (b,disp,h,w); computes all 16 groups (sum over
// 8 channels each, channel index c = cpg*16 + g), leaky-ReLU, bf16 pack,
// one fully coalesced 32B store per thread.
// ---------------------------------------------------------------------------
__global__ __launch_bounds__(64)
void corr_kernel(const float* __restrict__ x, const float* __restrict__ nbx,
                 unsigned short* __restrict__ costbf, int dil) {
  const int w   = threadIdx.x;
  const int lin = blockIdx.x;            // (b*81 + disp)*64 + h
  const int h   = lin & 63;
  int t         = lin >> 6;
  const int disp = t % ND;
  const int b    = t / ND;
  const int di = disp / 9 - 4;
  const int dj = disp % 9 - 4;
  const int h2 = h + di * dil;
  const int w2 = w + dj * dil;

  float acc[NG];
#pragma unroll
  for (int g = 0; g < NG; ++g) acc[g] = 0.f;

  if ((unsigned)h2 < (unsigned)NH && (unsigned)w2 < (unsigned)NW) {
    const float* x1p = x   + (size_t)b * NC * NH * NW + h  * NW + w;
    const float* x2p = nbx + (size_t)b * NC * NH * NW + h2 * NW + w2;
#pragma unroll 8
    for (int c = 0; c < NC; ++c) {
      acc[c & 15] = fmaf(x1p[(size_t)c * (NH * NW)],
                         x2p[(size_t)c * (NH * NW)], acc[c & 15]);
    }
  }

  unsigned pk[8];
#pragma unroll
  for (int g = 0; g < 8; ++g) {
    float a0 = acc[2 * g], a1 = acc[2 * g + 1];
    a0 = a0 > 0.f ? a0 : 0.1f * a0;              // leaky_relu(0.1)
    a1 = a1 > 0.f ? a1 : 0.1f * a1;
    pk[g] = (unsigned)f2bf(a0) | ((unsigned)f2bf(a1) << 16);
  }
  uint4* dst = (uint4*)(costbf + ((size_t)lin * NW + w) * NG);
  dst[0] = make_uint4(pk[0], pk[1], pk[2], pk[3]);
  dst[1] = make_uint4(pk[4], pk[5], pk[6], pk[7]);
}

// ---------------------------------------------------------------------------
// Weight swizzle into WMMA 16-bit A-matrix per-lane fragment layout.
// afrag[dil][kd][kh][slice][lane][16 bf16]; logical K = kw_local*16 + g_in,
// kw_local = 2*slice + (K>=16), kw_local==3 -> zero pad.
// A-layout: lane: M=lane&15, half=lane>>4; element e (v=e>>1):
//   K(e) = 16*half + e
// ---------------------------------------------------------------------------
__global__ __launch_bounds__(128)
void prep_kernel(const float* __restrict__ conv_w,
                 unsigned short* __restrict__ afrag) {
  int t = blockIdx.x * blockDim.x + threadIdx.x;   // 4*3*3*2*32 = 2304 lanes
  if (t >= 4 * 3 * 3 * 2 * 32) return;
  const int lane = t & 31;
  int r = t >> 5;
  const int s  = r & 1;  r >>= 1;
  const int kh = r % 3;  r /= 3;
  const int kd = r % 3;  r /= 3;
  const int dil = r;
  const int M = lane & 15, half = lane >> 4;

  unsigned pk[8];
#pragma unroll
  for (int p = 0; p < 8; ++p) {
    unsigned short two[2];
#pragma unroll
    for (int bit = 0; bit < 2; ++bit) {
      const int e = 2 * p + bit;
      const int v = e >> 1;
      const int K = 16 * half + e;
      const int kw = 2 * s + (K >> 4);
      const int gi = K & 15;
      float wv = 0.f;
      if (kw < 3)
        wv = conv_w[(((((size_t)dil * NG + M) * NG + gi) * 3 + kd) * 3 + kh) * 3 + kw];
      two[bit] = f2bf(wv);
    }
    pk[p] = (unsigned)two[0] | ((unsigned)two[1] << 16);
  }
  uint4* dst = (uint4*)(afrag + (size_t)t * 16);
  dst[0] = make_uint4(pk[0], pk[1], pk[2], pk[3]);
  dst[1] = make_uint4(pk[4], pk[5], pk[6], pk[7]);
}

// ---------------------------------------------------------------------------
// Conv3d 3x3x3 as WMMA GEMM. Workgroup = 128 threads (4 waves) handles one
// (b, d, h) row; each wave owns a 16-wide w tile (M=16 out-ch x N=16 w).
// LDS stages the halo tile pixel-major: sC[kd][kh][wwi][g16], wwi = ww+1.
// In-range 16B chunks are copied global->LDS with the CDNA5 async copier
// (GLOBAL_LOAD_ASYNC_TO_LDS_B128, tracked by ASYNCcnt); out-of-range halo
// chunks are zero-filled with plain LDS stores. s_wait_asynccnt 0 + barrier
// makes the tile visible before the WMMA loop.
// B-fragment per lane (N=lane&15, half=lane>>4) for slice s:
//   elements = the 16 g values of pixel (w0+N + 2s + half)  -> 2x ds_load_b128
// 18 wmma per tile; fused (bias,BN,ReLU) epilogue on f32 accumulators.
// ---------------------------------------------------------------------------
__global__ __launch_bounds__(128)
void conv_kernel(const unsigned short* __restrict__ costbf,
                 const unsigned short* __restrict__ afrag,
                 const float* __restrict__ conv_b,
                 const float* __restrict__ bn_gamma,
                 const float* __restrict__ bn_beta,
                 const float* __restrict__ bn_mean,
                 const float* __restrict__ bn_var,
                 float* __restrict__ out, int dil_idx) {
  __shared__ __align__(16) unsigned short sC[3][3][68][NG];

  const int lin = blockIdx.x;            // (b*81 + d)*64 + h
  const int h   = lin & 63;
  int t         = lin >> 6;
  const int d   = t % ND;
  const int b   = t / ND;
  const int tid = threadIdx.x;

  // --- cooperative LDS fill: async copy for in-range, ds zero for halo ---
  for (int chunk = tid; chunk < 3 * 3 * 68 * 2; chunk += 128) {
    const int hs = chunk & 1;
    int p = chunk >> 1;
    const int wwi = p % 68; p /= 68;
    const int kh = p % 3;
    const int kd = p / 3;
    const int dd = d - 1 + kd;
    const int hh = h - 1 + kh;
    const int ww = wwi - 1;
    unsigned short* lptr = &sC[kd][kh][wwi][hs * 8];
    if ((unsigned)dd < (unsigned)ND && (unsigned)hh < (unsigned)NH &&
        (unsigned)ww < (unsigned)NW) {
      const unsigned short* gptr =
          costbf + ((((size_t)b * ND + dd) * NH + hh) * NW + ww) * NG + hs * 8;
      unsigned lds_addr = (unsigned)(size_t)lptr;   // LDS byte address
      asm volatile("global_load_async_to_lds_b128 %0, %1, off"
                   :
                   : "v"(lds_addr), "v"(gptr)
                   : "memory");
    } else {
      *(uint4*)lptr = make_uint4(0u, 0u, 0u, 0u);
    }
  }
  asm volatile("s_wait_asynccnt 0x0" ::: "memory");
  __syncthreads();

  const int wv   = tid >> 5;
  const int lane = tid & 31;
  const int N    = lane & 15;
  const int half = lane >> 4;
  const int w0   = wv * 16;

  v8f acc = {0.f, 0.f, 0.f, 0.f, 0.f, 0.f, 0.f, 0.f};
  const unsigned short* abase = afrag + (size_t)dil_idx * (3 * 3 * 2 * 32 * 16);

#pragma unroll
  for (int kd = 0; kd < 3; ++kd) {
#pragma unroll
    for (int kh = 0; kh < 3; ++kh) {
#pragma unroll
      for (int s = 0; s < 2; ++s) {
        Frag32 af, bf;
        const uint4* ap =
            (const uint4*)(abase + ((((kd * 3 + kh) * 2 + s) * 32 + lane)) * 16);
        af.q[0] = ap[0];
        af.q[1] = ap[1];
        const uint4* bp = (const uint4*)(&sC[kd][kh][w0 + N + 2 * s + half][0]);
        bf.q[0] = bp[0];
        bf.q[1] = bp[1];
        acc = __builtin_amdgcn_wmma_f32_16x16x32_bf16(
            false, af.v, false, bf.v, (short)0, acc, false, false);
      }
    }
  }

  // --- fused bias + BN(eval) + ReLU epilogue; C/D layout: elem r -> M=r+8*half
#pragma unroll
  for (int r = 0; r < 8; ++r) {
    const int M  = r + 8 * half;
    const int ci = dil_idx * NG + M;
    const float inv  = bn_gamma[ci] * rsqrtf(bn_var[ci] + 1e-5f);
    const float bias = bn_beta[ci] + (conv_b[ci] - bn_mean[ci]) * inv;
    float o = acc[r] * inv + bias;
    o = fmaxf(o, 0.f);
    // Output written once, never re-read -> non-temporal to preserve L2 for
    // the cost volume + inputs.
    __builtin_nontemporal_store(
        o, &out[((((size_t)b * 64 + ci) * ND + d) * NH + h) * NW + (w0 + N)]);
  }
}

// ---------------------------------------------------------------------------
extern "C" void kernel_launch(void* const* d_in, const int* in_sizes, int n_in,
                              void* d_out, int out_size, void* d_ws, size_t ws_size,
                              hipStream_t stream) {
  const float* x        = (const float*)d_in[0];
  const float* nbx      = (const float*)d_in[1];
  const float* conv_w   = (const float*)d_in[2];
  const float* conv_b   = (const float*)d_in[3];
  const float* bn_gamma = (const float*)d_in[4];
  const float* bn_beta  = (const float*)d_in[5];
  const float* bn_mean  = (const float*)d_in[6];
  const float* bn_var   = (const float*)d_in[7];
  float* out = (float*)d_out;

  // ws carve: [0,72KB) swizzled A fragments, [128KB, +20.25MB) bf16 cost volume
  unsigned short* afrag  = (unsigned short*)d_ws;
  unsigned short* costbf = (unsigned short*)((char*)d_ws + (128u * 1024u));

  prep_kernel<<<(4 * 3 * 3 * 2 * 32 + 127) / 128, 128, 0, stream>>>(conv_w, afrag);

  const int dils[4] = {1, 2, 4, 8};
  const int nblk = NB * ND * NH;   // 10368
  for (int di = 0; di < 4; ++di) {
    corr_kernel<<<nblk, 64, 0, stream>>>(x, nbx, costbf, dils[di]);
    conv_kernel<<<nblk, 128, 0, stream>>>(costbf, afrag, conv_b, bn_gamma,
                                          bn_beta, bn_mean, bn_var, out, di);
  }
}
